// PauliNetEmbedding_42374147342438
// MI455X (gfx1250) — compile-verified, matrix operand order checked by reading, not run on hardware
//
#include <hip/hip_runtime.h>
#include <hip/hip_bf16.h>

typedef __bf16 bf16_t;
typedef __attribute__((ext_vector_type(16))) __bf16 v16bf;
typedef __attribute__((ext_vector_type(8)))  float  v8f;

#define LDSS 72   // LDS row stride in bf16 elems (144B: 16B-aligned, bank-conflict-free)

__device__ inline bf16_t f2bf(float f) { return (bf16_t)f; }

// Branch-free tanh using the hardware transcendental units:
// tanh(x) = 1 - 2/(exp(2x)+1); exp(2x) = exp2(x * 2*log2(e)).
// Saturates naturally: exp2->inf => 1, exp2->0 => -1. No EXEC divergence.
__device__ inline float fast_tanh(float x) {
  float e = __builtin_amdgcn_exp2f(x * 2.8853900817779268f);
  return 1.0f - 2.0f * __builtin_amdgcn_rcpf(e + 1.0f);
}

// ---------------------------------------------------------------------------
// One MLP layer worth of WMMA: Kpad = 32*KT, Npad = 64 (4 N-tiles).
// kt-outer / nt-inner: A fragment loaded once per kt (2x ds_load_b128), four
// independent accumulator chains so WMMAs pipeline without RAW stalls.
// A layout (16x32 bf16): lane L -> row M=L%16, K base (L>=16 ? 8 : 0);
// elements 0..7 = K base..base+7, elements 8..15 = K base+16..base+23.
// B fragments pre-packed per-lane-contiguous by k_pack_layer (mirrored layout).
// ---------------------------------------------------------------------------
template<int KT>
__device__ inline void layer_mma(const bf16_t* __restrict__ src,
                                 const bf16_t* __restrict__ pw,
                                 int lane, v8f c[4]) {
  const int half = lane >> 4;
  const int m    = lane & 15;
#pragma unroll
  for (int kt = 0; kt < KT; ++kt) {
    union { v16bf v; uint4 q[2]; } a;
    const bf16_t* ap = src + m * LDSS + kt * 32 + (half << 3);
    a.q[0] = *(const uint4*)(ap);
    a.q[1] = *(const uint4*)(ap + 16);
#pragma unroll
    for (int nt = 0; nt < 4; ++nt) {
      union { v16bf v; uint4 q[2]; } b;
      const bf16_t* bp = pw + (size_t)((nt * KT + kt) * 512 + lane * 16);
      b.q[0] = *(const uint4*)(bp);
      b.q[1] = *(const uint4*)(bp + 8);
      c[nt] = __builtin_amdgcn_wmma_f32_16x16x32_bf16(false, a.v, false, b.v,
                                                      (short)0, c[nt], false, false);
    }
  }
}

template<int KT, bool TANH>
__device__ inline void layer_to_lds(const bf16_t* __restrict__ src,
                                    bf16_t* __restrict__ dst,
                                    const bf16_t* __restrict__ pw,
                                    const float* __restrict__ pb,
                                    int lane) {
  const int half = lane >> 4;
  const int nlo  = lane & 15;
  v8f c[4] = {{}, {}, {}, {}};
  layer_mma<KT>(src, pw, lane, c);
#pragma unroll
  for (int nt = 0; nt < 4; ++nt) {
    float bias = pb[nt * 16 + nlo];
#pragma unroll
    for (int v = 0; v < 8; ++v) {
      float x = c[nt][v] + bias;
      if (TANH) x = fast_tanh(x);
      dst[(v + (half << 3)) * LDSS + nt * 16 + nlo] = f2bf(x);
    }
  }
}

// ---------------------------------------------------------------------------
// Fused MLP kernel: 8 waves/block, each wave owns a 16-row tile through all
// layers (rows are multiples of 128, so EXEC stays all-ones for WMMA).
// EPI: 0=store f32 (rows x 64); 1=pair-same; 2=pair-diff; 3=ion;
// 4=g (f32 to d_out h_el + bf16 act for next iteration).
// ---------------------------------------------------------------------------
template<int NL, int EPI, bool LTANH, int K0T>
__global__ __launch_bounds__(256) void mlp_kernel(
    const bf16_t* __restrict__ actIn, int inStride,
    const bf16_t* __restrict__ pw0, const float* __restrict__ pb0,
    const bf16_t* __restrict__ pw1, const float* __restrict__ pb1,
    const bf16_t* __restrict__ pw2, const float* __restrict__ pb2,
    float* __restrict__ outF, const float* __restrict__ hfac,
    bf16_t* __restrict__ actOut)
{
  __shared__ __align__(16) bf16_t lds[2][8][16 * LDSS];
  const int wave = threadIdx.x >> 5;
  const int lane = threadIdx.x & 31;
  const int half = lane >> 4;
  const int nlo  = lane & 15;
  const int rowTile = blockIdx.x * 128 + wave * 16;
  bf16_t* b0 = lds[0][wave];
  bf16_t* b1 = lds[1][wave];

  // Stage this wave's 16-row input tile into LDS (16B chunks).
  constexpr int CPR = K0T * 4;
#pragma unroll
  for (int idx = 0; idx < (16 * CPR + 31) / 32; ++idx) {
    int flat = idx * 32 + lane;
    int r = flat / CPR, ch = flat % CPR;
    if (16 * CPR % 32 == 0 || flat < 16 * CPR) {
      uint4 v = *(const uint4*)(actIn + (size_t)(rowTile + r) * inStride + ch * 8);
      *(uint4*)&b0[r * LDSS + ch * 8] = v;
    }
  }
  // Same-wave LDS ops are in-order; no barrier required.

  const bf16_t* src = b0;
  if (NL >= 2) { layer_to_lds<K0T, true>(b0, b1, pw0, pb0, lane); src = b1; }
  if (NL >= 3) { layer_to_lds<2,   true>(b1, b0, pw1, pb1, lane); src = b0; }
  const bf16_t* pwL = (NL == 1) ? pw0 : (NL == 2) ? pw1 : pw2;
  const float*  pbL = (NL == 1) ? pb0 : (NL == 2) ? pb1 : pb2;
  constexpr int KTL = (NL == 1) ? K0T : 2;

  v8f c[4] = {{}, {}, {}, {}};
  layer_mma<KTL>(src, pwL, lane, c);

#pragma unroll
  for (int nt = 0; nt < 4; ++nt) {
    float bias = pbL[nt * 16 + nlo];
    int col = nt * 16 + nlo;
#pragma unroll
    for (int v = 0; v < 8; ++v) {
      float x = c[nt][v] + bias;
      if (LTANH) x = fast_tanh(x);
      int row = rowTile + v + (half << 3);
      if constexpr (EPI == 0) {
        outF[(size_t)row * 64 + col] = x;
      } else if constexpr (EPI == 1 || EPI == 2) {
        int b = row >> 7, r = row & 127, i, j;
        if constexpr (EPI == 1) {           // same-spin: [uu ; dd]
          if (r < 64) { i = r >> 3; j = r & 7; }
          else { int u = r - 64; i = 8 + (u >> 3); j = 8 + (u & 7); }
        } else {                            // diff-spin: [ud ; du]
          if (r < 64) { i = r >> 3; j = 8 + (r & 7); }
          else { int u = r - 64; i = 8 + (u >> 3); j = u & 7; }
        }
        float h = hfac[((size_t)b * 16 + j) * 64 + col];
        outF[(((size_t)b * 16 + i) * 16 + j) * 64 + col] = x * h;
      } else if constexpr (EPI == 3) {
        int b = row >> 6, r = row & 63, i = r >> 2, m = r & 3;
        float h = hfac[((size_t)b * 4 + m) * 64 + col];
        outF[(((size_t)b * 16 + i) * 4 + m) * 64 + col] = x * h;
      } else {                              // EPI == 4 (g output)
        outF[(size_t)row * 64 + col] = x;
        actOut[(size_t)row * 64 + col] = f2bf(x);
      }
    }
  }
}

// ---------------------------------------------------------------------------
// Weight packer: f32 (K,N) row-major -> bf16 B-operand fragments, padded to
// Kpad (32*ktiles) x 64. Fragment f = nt*ktiles+kt, per-lane 16 contiguous
// bf16 (element s -> VGPR s/2, half s%2). Bias padded to 64 f32.
// ---------------------------------------------------------------------------
__global__ void k_pack_layer(const float* __restrict__ W, const float* __restrict__ bsrc,
                             bf16_t* __restrict__ pw, float* __restrict__ pb,
                             int K, int N, int ktiles) {
  int tot = ktiles * 4 * 512;
  for (int idx = threadIdx.x; idx < tot; idx += blockDim.x) {
    int f = idx >> 9, e = idx & 511;
    int lane = e >> 4, s = e & 15;
    int nt = f / ktiles, kt = f % ktiles;
    int v = s >> 1, h = s & 1;
    int kin = (v < 4 ? 2 * v + h : 16 + 2 * (v - 4) + h) + ((lane >= 16) ? 8 : 0);
    int k = kt * 32 + kin;
    int n = nt * 16 + (lane & 15);
    float val = (k < K && n < N) ? W[k * N + n] : 0.0f;
    pw[idx] = f2bf(val);
  }
  if (threadIdx.x < 64) pb[threadIdx.x] = (threadIdx.x < (unsigned)N) ? bsrc[threadIdx.x] : 0.0f;
}

// ---------------------------------------------------------------------------
// Input builders / elementwise kernels
// ---------------------------------------------------------------------------
__global__ void k_build_pairs(const float* __restrict__ elel,
                              bf16_t* __restrict__ same, bf16_t* __restrict__ diff) {
  size_t t = (size_t)blockIdx.x * blockDim.x + threadIdx.x;   // B*128*32
  int f = (int)(t & 31);
  size_t rr = t >> 5;
  int r = (int)(rr & 127);
  size_t b = rr >> 7;
  int iS, jS, iD, jD;
  if (r < 64) { iS = r >> 3; jS = r & 7; iD = r >> 3; jD = 8 + (r & 7); }
  else { int u = r - 64; iS = 8 + (u >> 3); jS = 8 + (u & 7); iD = 8 + (u >> 3); jD = u & 7; }
  same[t] = f2bf(elel[(((b * 16 + iS) * 16) + jS) * 32 + f]);
  diff[t] = f2bf(elel[(((b * 16 + iD) * 16) + jD) * 32 + f]);
}

__global__ void k_cast_bf16(const float* __restrict__ src, bf16_t* __restrict__ dst) {
  size_t t = (size_t)blockIdx.x * blockDim.x + threadIdx.x;
  dst[t] = f2bf(src[t]);
}

__global__ void k_build_pad(const float* __restrict__ src, bf16_t* __restrict__ dst, int cols) {
  size_t t = (size_t)blockIdx.x * blockDim.x + threadIdx.x;   // rows*64
  int c = (int)(t & 63);
  size_t row = t >> 6;
  dst[t] = (c < cols) ? f2bf(src[row * cols + c]) : f2bf(0.0f);
}

__global__ void k_copy_f32(const float* __restrict__ src, float* __restrict__ dst) {
  size_t t = (size_t)blockIdx.x * blockDim.x + threadIdx.x;
  dst[t] = src[t];
}

__global__ void k_sum_rows(const float* __restrict__ elel, const float* __restrict__ elionf,
                           bf16_t* __restrict__ actg) {
  size_t t = (size_t)blockIdx.x * blockDim.x + threadIdx.x;   // B*16*64
  int c = (int)(t & 63);
  size_t row = t >> 6;                                         // b*16+i
  float s = 0.0f;
  const float* pe = elel + (row * 16) * 64 + c;
#pragma unroll
  for (int j = 0; j < 16; ++j) s += pe[(size_t)j * 64];
  const float* pi = elionf + (row * 4) * 64 + c;
#pragma unroll
  for (int m = 0; m < 4; ++m) s += pi[(size_t)m * 64];
  actg[t] = f2bf(s);
}

// ---------------------------------------------------------------------------
// Host orchestration
// ---------------------------------------------------------------------------
extern "C" void kernel_launch(void* const* d_in, const int* in_sizes, int n_in,
                              void* d_out, int out_size, void* d_ws, size_t ws_size,
                              hipStream_t stream) {
  (void)n_in; (void)out_size; (void)ws_size;
  const int B = 2048;
  const float* f_elel  = (const float*)d_in[0];
  const float* f_el    = (const float*)d_in[1];
  const float* f_ion   = (const float*)d_in[2];
  const float* f_elion = (const float*)d_in[3];

  // Param leaf indexing: support insertion order and alphabetically-sorted
  // flattening (d_in[4] is h_ion_map W (32*64=2048) vs g_0 W (64*40=2560)).
  int I_ionmap, I_hsame[3], I_hdiff[3], I_wsame[3], I_wdiff[3], I_wion[3], I_g[3];
  if (in_sizes[4] == 2048) {
    I_ionmap = 4;
    for (int n = 0; n < 3; ++n) {
      int s = 6 + n * 34;
      I_hsame[n] = s; I_hdiff[n] = s + 6; I_wsame[n] = s + 12;
      I_wdiff[n] = s + 18; I_wion[n] = s + 24; I_g[n] = s + 30;
    }
  } else {  // sorted: g_0..2, h_diff_0..2, h_ion_map, h_same_0..2, w_diff, w_ion, w_same
    I_g[0] = 4; I_g[1] = 8; I_g[2] = 12;
    I_hdiff[0] = 16; I_hdiff[1] = 22; I_hdiff[2] = 28;
    I_ionmap = 34;
    I_hsame[0] = 36; I_hsame[1] = 42; I_hsame[2] = 48;
    I_wdiff[0] = 54; I_wdiff[1] = 60; I_wdiff[2] = 66;
    I_wion[0] = 72; I_wion[1] = 78; I_wion[2] = 84;
    I_wsame[0] = 90; I_wsame[1] = 96; I_wsame[2] = 102;
  }
  auto P = [&](int i) { return (const float*)d_in[i]; };

  // Workspace carve-out
  char* wsb = (char*)d_ws;
  size_t off = 0;
  auto take = [&](size_t bytes) -> char* {
    char* p = wsb + off; off = (off + bytes + 255) & ~(size_t)255; return p;
  };
  bf16_t* pw[52]; float* pb[52];
  for (int s = 0; s < 52; ++s) { pw[s] = (bf16_t*)take(64 * 64 * 2); pb[s] = (float*)take(64 * 4); }
  bf16_t* act_same  = (bf16_t*)take((size_t)B * 128 * 32 * 2);
  bf16_t* act_diff  = (bf16_t*)take((size_t)B * 128 * 32 * 2);
  bf16_t* act_elion = (bf16_t*)take((size_t)B * 64 * 32 * 2);
  bf16_t* act_h     = (bf16_t*)take((size_t)B * 16 * 64 * 2);
  bf16_t* act_g     = (bf16_t*)take((size_t)B * 16 * 64 * 2);
  bf16_t* act_ion   = (bf16_t*)take((size_t)B * 4 * 64 * 2);
  float*  h_same    = (float*)take((size_t)B * 16 * 64 * 4);
  float*  h_diff    = (float*)take((size_t)B * 16 * 64 * 4);
  float*  h_ionm    = (float*)take((size_t)B * 4 * 64 * 4);

  // Output regions: h_el | h_ion | h_el_el | h_el_ion_feat
  float* out      = (float*)d_out;
  float* o_hel    = out;
  float* o_hion   = out + (size_t)B * 16 * 64;
  float* o_elel   = o_hion + (size_t)B * 4 * 32;
  float* o_elionf = o_elel + (size_t)B * 16 * 16 * 64;

  // Pack weights (slot order: ionmap; per iter: h_same x3, h_diff x3, w_same x3,
  // w_diff x3, w_ion x3, g x2)
  int slot = 0;
  auto pack = [&](int wIdx, int K, int N) {
    int kt = (K + 31) / 32;
    k_pack_layer<<<1, 256, 0, stream>>>(P(wIdx), P(wIdx + 1), pw[slot], pb[slot], K, N, kt);
    ++slot;
  };
  pack(I_ionmap, 32, 64);
  for (int n = 0; n < 3; ++n) {
    int dinH = (n == 0) ? 32 : 64;
    pack(I_hsame[n] + 0, dinH, 40); pack(I_hsame[n] + 2, 40, 40); pack(I_hsame[n] + 4, 40, 64);
    pack(I_hdiff[n] + 0, dinH, 40); pack(I_hdiff[n] + 2, 40, 40); pack(I_hdiff[n] + 4, 40, 64);
    pack(I_wsame[n] + 0, 32, 40);   pack(I_wsame[n] + 2, 40, 40); pack(I_wsame[n] + 4, 40, 64);
    pack(I_wdiff[n] + 0, 32, 40);   pack(I_wdiff[n] + 2, 40, 40); pack(I_wdiff[n] + 4, 40, 64);
    pack(I_wion[n]  + 0, 32, 40);   pack(I_wion[n]  + 2, 40, 40); pack(I_wion[n]  + 4, 40, 64);
    pack(I_g[n] + 0, 64, 40);       pack(I_g[n] + 2, 40, 64);
  }

  // Build bf16 activations + copy h_ion passthrough
  k_build_pairs<<<(unsigned)(((size_t)B * 128 * 32) / 256), 256, 0, stream>>>(f_elel, act_same, act_diff);
  k_cast_bf16  <<<(unsigned)(((size_t)B * 64 * 32) / 256), 256, 0, stream>>>(f_elion, act_elion);
  k_build_pad  <<<(unsigned)(((size_t)B * 16 * 64) / 256), 256, 0, stream>>>(f_el, act_h, 32);
  k_build_pad  <<<(unsigned)(((size_t)B * 4 * 64) / 256), 256, 0, stream>>>(f_ion, act_ion, 32);
  k_copy_f32   <<<(unsigned)(((size_t)B * 4 * 32) / 256), 256, 0, stream>>>(f_ion, o_hion);

  // h_ion_mapped = tanh(h_ion @ W + b)   (B*4 rows, 32 -> 64)
  mlp_kernel<1, 0, true, 1><<<(B * 4) / 128, 256, 0, stream>>>(
      act_ion, 64, pw[0], pb[0], nullptr, nullptr, nullptr, nullptr,
      h_ionm, nullptr, nullptr);

  for (int n = 0; n < 3; ++n) {
    int bs = 1 + n * 17;
    // h_same / h_diff (B*16 rows); iter 0 has K0=32, later iters K0=64
    if (n == 0) {
      mlp_kernel<3, 0, true, 1><<<(B * 16) / 128, 256, 0, stream>>>(
          act_h, 64, pw[bs + 0], pb[bs + 0], pw[bs + 1], pb[bs + 1], pw[bs + 2], pb[bs + 2],
          h_same, nullptr, nullptr);
      mlp_kernel<3, 0, true, 1><<<(B * 16) / 128, 256, 0, stream>>>(
          act_h, 64, pw[bs + 3], pb[bs + 3], pw[bs + 4], pb[bs + 4], pw[bs + 5], pb[bs + 5],
          h_diff, nullptr, nullptr);
    } else {
      mlp_kernel<3, 0, true, 2><<<(B * 16) / 128, 256, 0, stream>>>(
          act_h, 64, pw[bs + 0], pb[bs + 0], pw[bs + 1], pb[bs + 1], pw[bs + 2], pb[bs + 2],
          h_same, nullptr, nullptr);
      mlp_kernel<3, 0, true, 2><<<(B * 16) / 128, 256, 0, stream>>>(
          act_h, 64, pw[bs + 3], pb[bs + 3], pw[bs + 4], pb[bs + 4], pw[bs + 5], pb[bs + 5],
          h_diff, nullptr, nullptr);
    }
    // w_same / w_diff fused with pair product -> h_el_el in d_out
    mlp_kernel<3, 1, false, 1><<<(B * 128) / 128, 256, 0, stream>>>(
        act_same, 32, pw[bs + 6], pb[bs + 6], pw[bs + 7], pb[bs + 7], pw[bs + 8], pb[bs + 8],
        o_elel, h_same, nullptr);
    mlp_kernel<3, 2, false, 1><<<(B * 128) / 128, 256, 0, stream>>>(
        act_diff, 32, pw[bs + 9], pb[bs + 9], pw[bs + 10], pb[bs + 10], pw[bs + 11], pb[bs + 11],
        o_elel, h_diff, nullptr);
    // w_ion fused with ion product -> h_el_ion_feat in d_out
    mlp_kernel<3, 3, false, 1><<<(B * 64) / 128, 256, 0, stream>>>(
        act_elion, 32, pw[bs + 12], pb[bs + 12], pw[bs + 13], pb[bs + 13], pw[bs + 14], pb[bs + 14],
        o_elionf, h_ionm, nullptr);
    // h_el_pre = sum_j h_el_el + sum_m h_el_ion_feat  (bf16 input for g)
    k_sum_rows<<<(unsigned)(((size_t)B * 16 * 64) / 256), 256, 0, stream>>>(o_elel, o_elionf, act_g);
    // g: writes f32 h_el (final iter value survives) + bf16 act for next iter
    mlp_kernel<2, 4, true, 2><<<(B * 16) / 128, 256, 0, stream>>>(
        act_g, 64, pw[bs + 15], pb[bs + 15], pw[bs + 16], pb[bs + 16], nullptr, nullptr,
        o_hel, nullptr, act_h);
  }
}
